// AttentionLayer_69380901699611
// MI455X (gfx1250) — matrix operand order repaired and checked by reading, not compiled, not run on hardware
//
#include <hip/hip_runtime.h>

#define HIDDEN 1024
#define HEADS  16
#define DEPTH  64
#define BATCH  2
#define SEQ    2048   // F == T == 2048

typedef __bf16 v16bf __attribute__((ext_vector_type(16)));
typedef __bf16 v8bf  __attribute__((ext_vector_type(8)));
typedef float  v8f   __attribute__((ext_vector_type(8)));

__device__ __forceinline__ v8f wmma_bf16(v16bf a, v16bf b, v8f c) {
  // D(f32 16x16) = A(bf16 16x32) * B(bf16 32x16) + C
  return __builtin_amdgcn_wmma_f32_16x16x32_bf16(false, a, false, b,
                                                 (short)0, c, false, false);
}

// --- DPP 16-lane row reductions (no LDS, co-issues with VALU) --------------
template <int CTRL>
__device__ __forceinline__ float dppf(float x) {
  return __builtin_bit_cast(float,
      __builtin_amdgcn_update_dpp(0, __builtin_bit_cast(int, x),
                                  CTRL, 0xF, 0xF, true));
}
// After quad_perm xor1+xor2 each quad is uniform, so the half-row and row
// mirrors correctly merge the 8- and 16-lane groups. Halves (lanes 0-15 vs
// 16-31) stay independent: DPP rows are 16 lanes.
__device__ __forceinline__ float rowmax16(float x) {
  x = fmaxf(x, dppf<0xB1>(x));   // quad_perm(1,0,3,2)  : xor 1
  x = fmaxf(x, dppf<0x4E>(x));   // quad_perm(2,3,0,1)  : xor 2
  x = fmaxf(x, dppf<0x141>(x));  // row_half_mirror
  x = fmaxf(x, dppf<0x140>(x));  // row_mirror
  return x;
}
__device__ __forceinline__ float rowsum16(float x) {
  x += dppf<0xB1>(x);
  x += dppf<0x4E>(x);
  x += dppf<0x141>(x);
  x += dppf<0x140>(x);
  return x;
}

// ---------------------------------------------------------------------------
// One-time weight prep: Wt[col][k] = (bf16)W[k][col]   (1024x1024)
// Tiled LDS transpose; coalesced on both sides. Runs once per weight (2MB out),
// so every GEMM B-fragment becomes a single contiguous 32B global load.
// ---------------------------------------------------------------------------
__global__ __launch_bounds__(256)
void transpose_cvt_kernel(const float* __restrict__ W, __bf16* __restrict__ Wt)
{
  __shared__ float tile[32][33];
  const int tx = threadIdx.x & 31;
  const int ty = threadIdx.x >> 5;       // 0..7
  const int i0 = blockIdx.y * 32;        // source row block  (k)
  const int j0 = blockIdx.x * 32;        // source col block  (col)
#pragma unroll
  for (int s = 0; s < 32; s += 8)
    tile[ty + s][tx] = W[(size_t)(i0 + ty + s) * HIDDEN + j0 + tx];
  __syncthreads();
#pragma unroll
  for (int s = 0; s < 32; s += 8)
    Wt[(size_t)(j0 + ty + s) * HIDDEN + i0 + tx] = (__bf16)tile[tx][ty + s];
}

// ---------------------------------------------------------------------------
// Projection GEMM: out = scale * X(f32 [4096,1024]) x W -> bf16
//   Wt: bf16 [1024 cols][1024 k]  (pre-transposed, k-contiguous)
//   MODE 0: out[((b*HEADS+n)*SEQ + s)*64 + h]   ([b,n,s,h]   : Q, K)
//   MODE 1: out[((b*HEADS+n)*64 + h)*SEQ + s]   ([b,n,h,s]   : V transposed)
// 128 threads = 4 waves; each wave computes 16 rows x 64 cols. No LDS, no
// barriers: B fragments are direct 32B global loads served from WGP$/L2.
// ---------------------------------------------------------------------------
template <int MODE>
__global__ __launch_bounds__(128)
void proj_gemm_kernel(const float* __restrict__ X,
                      const __bf16* __restrict__ Wt,
                      __bf16* __restrict__ out,
                      float scale)
{
  const int lane = threadIdx.x & 31;
  const int wave = threadIdx.x >> 5;
  const int half = lane >> 4;
  const int r    = lane & 15;

  const int rowBase = blockIdx.x * 64 + wave * 16;
  const int colBase = blockIdx.y * 64;

  v8f acc[4];
#pragma unroll
  for (int i = 0; i < 4; ++i)
#pragma unroll
    for (int j = 0; j < 8; ++j) acc[i][j] = 0.0f;

  const float* xrow = X + (size_t)(rowBase + r) * HIDDEN;

  for (int k0 = 0; k0 < HIDDEN; k0 += 32) {
    // A fragment: 16x32, row = r, k(e) = e + half*16
    const float* xr = xrow + k0;
    v16bf a;
#pragma unroll
    for (int e = 0; e < 8; ++e) {
      a[e]     = (__bf16)xr[e + half * 16];
      a[8 + e] = (__bf16)xr[8 + e + half * 16];
    }

#pragma unroll
    for (int nt = 0; nt < 4; ++nt) {
      // B fragment: 32x16, col = r, k(e) = e + half*16 -> contiguous in Wt
      const __bf16* wr =
          Wt + (size_t)(colBase + nt * 16 + r) * HIDDEN + k0 + half * 16;
      v16bf bf = *(const v16bf*)wr;     // one 32B load
      acc[nt] = wmma_bf16(a, bf, acc[nt]);
    }
  }

  // C layout: element j -> row = j + half*8, col = r (within 16x16 tile)
#pragma unroll
  for (int nt = 0; nt < 4; ++nt) {
#pragma unroll
    for (int j = 0; j < 8; ++j) {
      int grow = rowBase + j + half * 8;       // 0..4095
      int b    = grow >> 11;                   // /2048
      int s    = grow & 2047;
      int c2   = colBase + nt * 16 + r;        // n*64 + h
      int n    = c2 >> 6;
      int h    = c2 & 63;
      float v  = acc[nt][j] * scale;
      size_t off;
      if (MODE == 0)
        off = (((size_t)(b * HEADS + n) * SEQ) + s) * 64 + h;
      else
        off = ((size_t)(b * HEADS + n) * 64 + h) * SEQ + s;
      out[off] = (__bf16)v;
    }
  }
}

// ---------------------------------------------------------------------------
// Flash attention: one wave per (b, head, 16-row f tile).
//   Qh [b,n,f,64] bf16 (pre-scaled), Kh [b,n,t,64] bf16, Vt [b,n,64,t] bf16,
//   bias f32 [F,T], attn out bf16 [b*F, 1024] (col = n*64+h).
// ---------------------------------------------------------------------------
__global__ __launch_bounds__(32)
void flash_attn_kernel(const __bf16* __restrict__ Qh,
                       const __bf16* __restrict__ Kh,
                       const __bf16* __restrict__ Vt,
                       const float*  __restrict__ bias,
                       __bf16* __restrict__ attn)
{
  // row stride 40 elements (80B) keeps the two 8-element A-fragment chunks
  // 16B-aligned for vector LDS reads.
  __shared__ __attribute__((aligned(16))) __bf16 sP[16][40];

  const int lane = threadIdx.x & 31;
  const int half = lane >> 4;
  const int r    = lane & 15;

  const int f0 = blockIdx.x * 16;
  const int n  = blockIdx.y;
  const int b  = blockIdx.z;
  const size_t bn = (size_t)(b * HEADS + n);

  const __bf16* Qbase = Qh + (bn * SEQ + f0) * 64;
  const __bf16* Kbase = Kh + bn * SEQ * 64;
  const __bf16* Vbase = Vt + bn * 64 * SEQ;

  // Q A-fragments (k over depth: a0 -> h 0..31, a1 -> h 32..63)
  v16bf a0, a1;
  {
    const __bf16* qr = Qbase + r * 64;
#pragma unroll
    for (int e = 0; e < 8; ++e) {
      a0[e]     = qr[e + half * 16];
      a0[8 + e] = qr[8 + e + half * 16];
      a1[e]     = qr[32 + e + half * 16];
      a1[8 + e] = qr[40 + e + half * 16];
    }
  }

  v8f o[4];
  float m[8], l[8];
#pragma unroll
  for (int t = 0; t < 4; ++t)
#pragma unroll
    for (int j = 0; j < 8; ++j) o[t][j] = 0.0f;
#pragma unroll
  for (int j = 0; j < 8; ++j) { m[j] = -__builtin_inff(); l[j] = 0.0f; }

  for (int tb = 0; tb < SEQ; tb += 32) {
    // prefetch next iteration's K rows + bias rows into cache
    if (tb + 32 < SEQ) {
      __builtin_prefetch(Kbase + (size_t)(tb + 32 + r) * 64, 0, 3);
      __builtin_prefetch(Kbase + (size_t)(tb + 48 + r) * 64, 0, 3);
      __builtin_prefetch(bias + (size_t)(f0 + r) * SEQ + tb + 32, 0, 3);
    }

    v8f S0, S1;
    {
      const __bf16* kr = Kbase + (size_t)(tb + r) * 64;
      v16bf b0, b1;
#pragma unroll
      for (int e = 0; e < 16; ++e) {
        b0[e] = kr[half * 16 + e];
        b1[e] = kr[32 + half * 16 + e];
      }
      v8f z;
#pragma unroll
      for (int j = 0; j < 8; ++j) z[j] = 0.0f;
      S0 = wmma_bf16(a0, b0, z);
      S0 = wmma_bf16(a1, b1, S0);
    }
    {
      const __bf16* kr = Kbase + (size_t)(tb + 16 + r) * 64;
      v16bf b0, b1;
#pragma unroll
      for (int e = 0; e < 16; ++e) {
        b0[e] = kr[half * 16 + e];
        b1[e] = kr[32 + half * 16 + e];
      }
      v8f z;
#pragma unroll
      for (int j = 0; j < 8; ++j) z[j] = 0.0f;
      S1 = wmma_bf16(a0, b0, z);
      S1 = wmma_bf16(a1, b1, S1);
    }

    // bias + online softmax (C layout: row = j + half*8, col = r)
#pragma unroll
    for (int j = 0; j < 8; ++j) {
      int fr = f0 + j + half * 8;
      S0[j] += bias[(size_t)fr * SEQ + tb + r];
      S1[j] += bias[(size_t)fr * SEQ + tb + 16 + r];

      float x  = rowmax16(fmaxf(S0[j], S1[j]));
      float nm = fmaxf(m[j], x);
      float corr = __expf(m[j] - nm);
      m[j] = nm;
      float p0 = __expf(S0[j] - nm);
      float p1 = __expf(S1[j] - nm);
      float s  = rowsum16(p0 + p1);
      l[j] = l[j] * corr + s;
#pragma unroll
      for (int ht = 0; ht < 4; ++ht) o[ht][j] *= corr;
      sP[j + half * 8][r]      = (__bf16)p0;
      sP[j + half * 8][16 + r] = (__bf16)p1;
    }
    __syncthreads();

    // Reload P in A-fragment layout: row = r, k(e) = e + half*16
    // -> two contiguous 16B vector LDS reads per lane.
    v8bf plo = *(const v8bf*)&sP[r][half * 16];
    v8bf phi = *(const v8bf*)&sP[r][8 + half * 16];
    v16bf pf;
#pragma unroll
    for (int e = 0; e < 8; ++e) { pf[e] = plo[e]; pf[8 + e] = phi[e]; }
    __syncthreads();

    // o += P(16x32) x V(32x16) per 16-wide depth tile
#pragma unroll
    for (int ht = 0; ht < 4; ++ht) {
      const __bf16* vr = Vbase + (size_t)(ht * 16 + r) * SEQ + tb + half * 16;
      v16bf vb;
#pragma unroll
      for (int e = 0; e < 16; ++e) vb[e] = vr[e];
      o[ht] = wmma_bf16(pf, vb, o[ht]);
    }
  }

  float inv[8];
#pragma unroll
  for (int j = 0; j < 8; ++j) inv[j] = 1.0f / l[j];

#pragma unroll
  for (int ht = 0; ht < 4; ++ht) {
#pragma unroll
    for (int j = 0; j < 8; ++j) {
      int fr = f0 + j + half * 8;
      size_t row = (size_t)b * SEQ + fr;
      attn[row * HIDDEN + n * 64 + ht * 16 + r] = (__bf16)(o[ht][j] * inv[j]);
    }
  }
}

// ---------------------------------------------------------------------------
// Output GEMM: out(f32 [4096,1024]) = A(bf16 [4096,1024]) x wo
//   Wt: bf16 [1024 e-cols][1024 k]  (pre-transposed wo, k-contiguous)
// ---------------------------------------------------------------------------
__global__ __launch_bounds__(128)
void out_gemm_kernel(const __bf16* __restrict__ A,
                     const __bf16* __restrict__ Wt,
                     float* __restrict__ out)
{
  const int lane = threadIdx.x & 31;
  const int wave = threadIdx.x >> 5;
  const int half = lane >> 4;
  const int r    = lane & 15;

  const int rowBase = blockIdx.x * 64 + wave * 16;
  const int colBase = blockIdx.y * 64;

  v8f acc[4];
#pragma unroll
  for (int i = 0; i < 4; ++i)
#pragma unroll
    for (int j = 0; j < 8; ++j) acc[i][j] = 0.0f;

  const __bf16* arow = A + (size_t)(rowBase + r) * HIDDEN;

  for (int k0 = 0; k0 < HIDDEN; k0 += 32) {
    const __bf16* ar = arow + k0;
    v16bf a;
#pragma unroll
    for (int e = 0; e < 8; ++e) {
      a[e]     = ar[e + half * 16];
      a[8 + e] = ar[8 + e + half * 16];
    }

#pragma unroll
    for (int nt = 0; nt < 4; ++nt) {
      const __bf16* wr =
          Wt + (size_t)(colBase + nt * 16 + r) * HIDDEN + k0 + half * 16;
      v16bf bf = *(const v16bf*)wr;
      acc[nt] = wmma_bf16(a, bf, acc[nt]);
    }
  }

#pragma unroll
  for (int nt = 0; nt < 4; ++nt) {
#pragma unroll
    for (int j = 0; j < 8; ++j) {
      size_t row = (size_t)(rowBase + j + half * 8);
      out[row * HIDDEN + colBase + nt * 16 + r] = acc[nt][j];
    }
  }
}

// ---------------------------------------------------------------------------
extern "C" void kernel_launch(void* const* d_in, const int* in_sizes, int n_in,
                              void* d_out, int out_size, void* d_ws, size_t ws_size,
                              hipStream_t stream) {
  const float* query  = (const float*)d_in[0];  // [2,2048,1024]
  const float* source = (const float*)d_in[1];  // [2,2048,1024]
  const float* bias   = (const float*)d_in[2];  // [1,1,2048,2048]
  const float* wq     = (const float*)d_in[3];  // [1024,16,64] == [1024,1024]
  const float* wk     = (const float*)d_in[4];
  const float* wv     = (const float*)d_in[5];
  const float* wo     = (const float*)d_in[6];  // [16,64,1024] == [1024,1024]
  float* out = (float*)d_out;

  const size_t PROJ_ELEMS = (size_t)BATCH * HEADS * SEQ * DEPTH;  // 4M bf16
  const size_t W_ELEMS    = (size_t)HIDDEN * HIDDEN;              // 1M bf16
  __bf16* Qh  = (__bf16*)d_ws;
  __bf16* Kh  = Qh + PROJ_ELEMS;
  __bf16* Vt  = Kh + PROJ_ELEMS;
  __bf16* At  = Vt + PROJ_ELEMS;   // attn output, [4096,1024] bf16
  __bf16* WqT = At + PROJ_ELEMS;   // transposed bf16 weights
  __bf16* WkT = WqT + W_ELEMS;
  __bf16* WvT = WkT + W_ELEMS;
  __bf16* WoT = WvT + W_ELEMS;

  // one-time weight transpose+convert (f32 [k][col] -> bf16 [col][k])
  dim3 tgrid(HIDDEN / 32, HIDDEN / 32);
  transpose_cvt_kernel<<<tgrid, 256, 0, stream>>>(wq, WqT);
  transpose_cvt_kernel<<<tgrid, 256, 0, stream>>>(wk, WkT);
  transpose_cvt_kernel<<<tgrid, 256, 0, stream>>>(wv, WvT);
  transpose_cvt_kernel<<<tgrid, 256, 0, stream>>>(wo, WoT);

  dim3 gblk(128, 1, 1);
  dim3 ggrid(BATCH * SEQ / 64, HIDDEN / 64, 1);   // (64, 16)

  proj_gemm_kernel<0><<<ggrid, gblk, 0, stream>>>(query,  WqT, Qh, 0.125f); // Q * depth^-0.5
  proj_gemm_kernel<0><<<ggrid, gblk, 0, stream>>>(source, WkT, Kh, 1.0f);
  proj_gemm_kernel<1><<<ggrid, gblk, 0, stream>>>(source, WvT, Vt, 1.0f);   // V transposed

  dim3 agrid(SEQ / 16, HEADS, BATCH);             // (128, 16, 2)
  flash_attn_kernel<<<agrid, dim3(32, 1, 1), 0, stream>>>(Qh, Kh, Vt, bias, At);

  out_gemm_kernel<<<ggrid, gblk, 0, stream>>>(At, WoT, out);
}
